// Decoder_34093450395751
// MI455X (gfx1250) — matrix operand order repaired and checked by reading, not compiled, not run on hardware
//
#include <hip/hip_runtime.h>
#include <hip/hip_bf16.h>
#include <math.h>

// ---------------------------------------------------------------------------
// MI455X (gfx1250) implementation, round 2.
//
// Roofline: ~0.95 GB mandatory HBM traffic -> ~41us floor @ 23.3 TB/s.
//  * VN-MLP chain (~103 GFLOP): fp32 WMMA (K=4) would need ~50M WMMA instrs
//    (~2x the memory floor in issue time). Use V_WMMA_F32_16X16X32_BF16
//    (K=32, fp32 accumulate): 8x fewer matrix instrs, 2x fewer weight bytes.
//    All elementwise math (vn_relu, layernorms, softmax) stays fp32.
//  * Fused per-16-item pipeline lives in 150KB LDS -> 2 blocks/WGP (4 w/SIMD).
//  * h_cc GEMM: fp32 WMMA, A-fragments register-resident per 16-row strip
//    (read ref_feat exactly once), We staged in pad-132 LDS (conflict-free).
// d_ws requirement: 1 MB (bf16 weight staging).
// ---------------------------------------------------------------------------

typedef float  v2f  __attribute__((ext_vector_type(2)));
typedef float  v8f  __attribute__((ext_vector_type(8)));
typedef __bf16 bf16;
typedef __bf16 v8bf  __attribute__((ext_vector_type(8)));
typedef __bf16 v16bf __attribute__((ext_vector_type(16)));

#define FCH   256
#define NB    16            // items per block (3*NB = 48 gemm columns)
#define COLS  48
#define CHP2  520           // bf16 activation stride: 512 + 8 pad (16B, bank-safe)
#define CHPF  260           // fp32 [col][ch] stride: 256 + 4 pad
#define KVST  780           // fp32 k/v stride per item: 256*3 + 12 pad
#define EPSV  1e-5f
#define WSTRIDE 132         // kernel B LDS We row stride (floats)

union FragU { struct { v8bf lo, hi; } p; v16bf v; };

// ---------------------------------------------------------------------------
// bf16 WMMA fragment layout used by this code (wave32, 16x16x32; A/B share one k order):
//  A 16x32: lane L holds M=L&15; K = {g*16..g*16+15}, g=L>>4
//  B 32x16: lane L holds N=L&15; K = {g*16..g*16+15}
//  C/D    : lane L holds N=L&15; M = g*8 + r
// ---------------------------------------------------------------------------
__device__ __forceinline__ v16bf load_frag_a(const bf16* wrow, int k0, int g8) {
  FragU u;
  u.p.lo = *(const v8bf*)(wrow + k0 + 2 * g8);
  u.p.hi = *(const v8bf*)(wrow + k0 + 2 * g8 + 8);
  return u.v;
}
__device__ __forceinline__ v16bf load_frag_b(const bf16* xcol, int k0) {
  FragU u;
  u.p.lo = *(const v8bf*)(xcol + k0);
  u.p.hi = *(const v8bf*)(xcol + k0 + 8);
  return u.v;
}

// Y[Och x 48] = Wb(Och x Kch bf16, global) * X(Kch x 48 bf16 [col][CHP2], LDS)
// Output: bf16 into Yl, or fp32 into Yf ([col][CHPF]) when Yf != nullptr.
__device__ __forceinline__ void gemm_bf16(const bf16* __restrict__ Wb, int Och, int Kch,
                                          const bf16* __restrict__ Xl,
                                          bf16* __restrict__ Yl, float* __restrict__ Yf,
                                          int wave, int lane) {
  const int l15 = lane & 15;
  const int g8  = (lane >> 4) << 3;
  const int g16 = g8 << 1;
  const int ntiles = (Och >> 4) * 3;
  for (int t = wave; t < ntiles; t += 8) {
    const int mt  = t / 3;
    const int ct  = t - mt * 3;
    const int col = ct * 16 + l15;
    const bf16* wrow = Wb + (size_t)(mt * 16 + l15) * Kch;
    const bf16* xcol = Xl + col * CHP2 + g16;
    v8f acc = {0.f, 0.f, 0.f, 0.f, 0.f, 0.f, 0.f, 0.f};
    for (int k0 = 0; k0 < Kch; k0 += 32) {
      v16bf a = load_frag_a(wrow, k0, g8);       // 2x global_load_b128
      v16bf b = load_frag_b(xcol, k0);           // 2x ds_load_b128
      acc = __builtin_amdgcn_wmma_f32_16x16x32_bf16(false, a, false, b,
                                                    (short)0, acc, false, false);
    }
    const int mbase = mt * 16 + g8;
    if (Yf) {
#pragma unroll
      for (int r = 0; r < 8; ++r) Yf[col * CHPF + mbase + r] = acc[r];
    } else {
      v8bf ob;
#pragma unroll
      for (int r = 0; r < 8; ++r) ob[r] = (bf16)acc[r];
      *(v8bf*)(Yl + col * CHP2 + mbase) = ob;    // ds_store_b128
    }
  }
}

// vn_relu over the 3-component axis, in place on bf16 activations.
__device__ __forceinline__ void vn_relu_bf(bf16* __restrict__ X,
                                           const bf16* __restrict__ D, int tid) {
  for (int p = tid; p < FCH * NB; p += 256) {
    const int c = p >> 4, nl = p & 15;
    const int cb = nl * 3;
    float x0 = (float)X[(cb + 0) * CHP2 + c];
    float x1 = (float)X[(cb + 1) * CHP2 + c];
    float x2 = (float)X[(cb + 2) * CHP2 + c];
    const float d0 = (float)D[(cb + 0) * CHP2 + c];
    const float d1 = (float)D[(cb + 1) * CHP2 + c];
    const float d2 = (float)D[(cb + 2) * CHP2 + c];
    const float dot = x0 * d0 + x1 * d1 + x2 * d2;
    if (dot < 0.f) {
      const float f = dot / (d0 * d0 + d1 * d1 + d2 * d2 + 1e-9f);
      X[(cb + 0) * CHP2 + c] = (bf16)(x0 - f * d0);
      X[(cb + 1) * CHP2 + c] = (bf16)(x1 - f * d1);
      X[(cb + 2) * CHP2 + c] = (bf16)(x2 - f * d2);
    }
  }
}

__device__ __forceinline__ void mat3(const float* __restrict__ W,
                                     float x0, float x1, float x2, float* y) {
  y[0] = W[0] * x0 + W[1] * x1 + W[2] * x2;
  y[1] = W[3] * x0 + W[4] * x1 + W[5] * x2;
  y[2] = W[6] * x0 + W[7] * x1 + W[8] * x2;
}

__device__ __forceinline__ void vnrelu1(float* t, const float* d) {
#pragma unroll
  for (int c = 0; c < 3; ++c) {
    const float dot = t[c] * d[c];
    if (dot < 0.f) t[c] = t[c] - dot / (d[c] * d[c] + 1e-9f) * d[c];
  }
}

// ---------------------------------------------------------------------------
// fp32 -> bf16 weight staging
// ---------------------------------------------------------------------------
__global__ void __launch_bounds__(256)
cvt_bf16_kernel(const float* __restrict__ src, bf16* __restrict__ dst, int n) {
  const int i = blockIdx.x * 256 + threadIdx.x;
  if (i < n) dst[i] = (bf16)src[i];
}

// ---------------------------------------------------------------------------
// Kernel A: fused VN-MLP + layernorm + attention + x_cc path, 16 items/block.
// LDS: 3 regions x 12480 floats + 64 ext = 150,016 B -> 2 blocks/WGP.
// ---------------------------------------------------------------------------
__global__ void __launch_bounds__(256)
vn_decoder_kernel(const float* __restrict__ Z_V, const float* __restrict__ B_x,
                  const bf16* __restrict__ Wff1b, const bf16* __restrict__ W1b,
                  const bf16* __restrict__ U1b,   const bf16* __restrict__ W2b,
                  const bf16* __restrict__ U2b,   const bf16* __restrict__ W3b_,
                  const float* __restrict__ g1,   const float* __restrict__ b1,
                  const float* __restrict__ Wq,   const float* __restrict__ bq,
                  const float* __restrict__ Wk,   const float* __restrict__ bk,
                  const float* __restrict__ Wv,   const float* __restrict__ bv,
                  const float* __restrict__ Wo,   const float* __restrict__ bo,
                  const float* __restrict__ W3a,  const float* __restrict__ W3bf32,
                  const float* __restrict__ U3b,  const float* __restrict__ W3c,
                  const float* __restrict__ U3c,  const float* __restrict__ W3d,
                  const float* __restrict__ g2,   const float* __restrict__ b2,
                  const int* __restrict__ fg_idx, float* __restrict__ xout) {
  extern __shared__ float smem[];
  float* R0f = smem;                 // 12480 f32 (= 24960 bf16)
  float* R1f = smem + 12480;
  float* R2f = smem + 24960;
  float* ext = smem + 37440;         // 64: Wc(0..8), mu(16..31), rstd(32..47)
  bf16*  R0h = (bf16*)R0f;
  bf16*  R1h = (bf16*)R1f;
  bf16*  R2h = (bf16*)R2f;

  const int tid  = threadIdx.x;
  const int lane = tid & 31;
  const int wave = tid >> 5;
  const int n0   = blockIdx.x * NB;

  // Wc = W3b @ W3a (3x3): no nonlinearity between them in the reference.
  if (tid < 9) {
    const int d = tid / 3, c = tid - 3 * (tid / 3);
    float s = 0.f;
    for (int o = 0; o < FCH; ++o) s += W3bf32[d * FCH + o] * W3a[o * 3 + c];
    ext[tid] = s;
  }

  // Load Z_V tile -> bf16 [col][CHP2]; global side fully coalesced.
  {
    const float* src = Z_V + (size_t)n0 * (FCH * 3);
    for (int idx = tid; idx < FCH * 3 * NB; idx += 256) {
      const int nl  = idx / (FCH * 3);
      const int rem = idx - nl * (FCH * 3);
      const int c   = rem / 3;
      const int i   = rem - c * 3;
      R0h[(nl * 3 + i) * CHP2 + c] = (bf16)src[idx];
    }
  }
  __syncthreads();

  // ---- VN-MLP chain (bf16 WMMA, fp32 accumulate, activations in LDS) ----
  gemm_bf16(Wff1b, 2 * FCH, FCH, R0h, R1h, nullptr, wave, lane); __syncthreads();
  gemm_bf16(W1b, FCH, 2 * FCH, R1h, R0h, nullptr, wave, lane);   __syncthreads();
  gemm_bf16(U1b, FCH, FCH, R0h, R2h, nullptr, wave, lane);       __syncthreads();
  vn_relu_bf(R0h, R2h, tid);                                     __syncthreads();
  gemm_bf16(W2b, FCH, FCH, R0h, R1h, nullptr, wave, lane);       __syncthreads();
  gemm_bf16(U2b, FCH, FCH, R1h, R2h, nullptr, wave, lane);       __syncthreads();
  vn_relu_bf(R1h, R2h, tid);                                     __syncthreads();
  gemm_bf16(W3b_, FCH, FCH, R1h, nullptr, R0f, wave, lane);      __syncthreads();

  // ---- residual (Z_V re-read: resident in 192MB L2) + vector layernorm ----
  for (int p = tid; p < FCH * NB; p += 256) {
    const int c = p >> 4, nl = p & 15;
    const float* zv = Z_V + (size_t)(n0 + nl) * (FCH * 3) + c * 3;
    float r[3];
    float nrm2 = 0.f;
#pragma unroll
    for (int i = 0; i < 3; ++i) {
      r[i] = R0f[(nl * 3 + i) * CHPF + c] + zv[i];
      R0f[(nl * 3 + i) * CHPF + c] = r[i];
      nrm2 += r[i] * r[i];
    }
    R2f[c * NB + nl] = sqrtf(nrm2);
  }
  __syncthreads();
  if (tid < NB) {
    float s = 0.f, s2 = 0.f;
    for (int c = 0; c < FCH; ++c) { const float v = R2f[c * NB + tid]; s += v; s2 += v * v; }
    const float mu  = s * (1.f / FCH);
    const float var = s2 * (1.f / FCH) - mu * mu;
    ext[16 + tid] = mu;
    ext[32 + tid] = 1.f / sqrtf(var + EPSV);
  }
  __syncthreads();
  for (int p = tid; p < FCH * NB; p += 256) {
    const int c = p >> 4, nl = p & 15;
    const float nrm = R2f[c * NB + nl];
    const float ln  = (nrm - ext[16 + nl]) * ext[32 + nl] * g1[c] + b1[c];
    const float sc  = ln / (nrm + EPSV);
#pragma unroll
    for (int i = 0; i < 3; ++i) R0f[(nl * 3 + i) * CHPF + c] *= sc;
  }
  __syncthreads();

  // ---- k, v for all 256 keys of each item: k -> R1f, v -> R2f ----
  {
    float wkr[9], wvr[9], bkr[3], bvr[3];
#pragma unroll
    for (int i = 0; i < 9; ++i) { wkr[i] = Wk[i]; wvr[i] = Wv[i]; }
#pragma unroll
    for (int i = 0; i < 3; ++i) { bkr[i] = bk[i]; bvr[i] = bv[i]; }
    for (int p = tid; p < FCH * NB; p += 256) {
      const int c = p >> 4, nl = p & 15;
      const float z0 = R0f[(nl * 3 + 0) * CHPF + c];
      const float z1 = R0f[(nl * 3 + 1) * CHPF + c];
      const float z2 = R0f[(nl * 3 + 2) * CHPF + c];
      float* kp_ = R1f + nl * KVST + c * 3;
      float* vp_ = R2f + nl * KVST + c * 3;
#pragma unroll
      for (int d = 0; d < 3; ++d) {
        kp_[d] = wkr[d * 3] * z0 + wkr[d * 3 + 1] * z1 + wkr[d * 3 + 2] * z2 + bkr[d];
        vp_[d] = wvr[d * 3] * z0 + wvr[d * 3 + 1] * z1 + wvr[d * 3 + 2] * z2 + bvr[d];
      }
    }
  }
  __syncthreads();

  // ---- attention (online softmax, whole wave shares one item -> LDS bcast) ----
  for (int task = tid; task < NB * 32; task += 256) {
    const int nl = task >> 5, j = task & 31;
    const int n = n0 + nl;
    const int cnt = (n & 31) + 1;                 // counts[n] = n % 32 + 1
    if (j >= cnt) continue;
    const int row = fg_idx[n * 32 + j];           // == off[n] + j == output row
    const float p0 = B_x[(size_t)row * 3 + 0];
    const float p1 = B_x[(size_t)row * 3 + 1];
    const float p2 = B_x[(size_t)row * 3 + 2];
    float q[3];
    mat3(Wq, p0, p1, p2, q);
    q[0] += bq[0]; q[1] += bq[1]; q[2] += bq[2];

    float m = -1e30f, l = 0.f, a0 = 0.f, a1 = 0.f, a2 = 0.f;
    const float* kb = R1f + nl * KVST;
    const float* vb = R2f + nl * KVST;
    for (int c = 0; c < FCH; ++c) {
      const float s = (q[0] * kb[c * 3] + q[1] * kb[c * 3 + 1] + q[2] * kb[c * 3 + 2])
                      * 0.57735026919f;
      const float mn   = fmaxf(m, s);
      const float corr = __expf(m - mn);
      const float w    = __expf(s - mn);
      a0 = a0 * corr + w * vb[c * 3];
      a1 = a1 * corr + w * vb[c * 3 + 1];
      a2 = a2 * corr + w * vb[c * 3 + 2];
      l  = l  * corr + w;
      m  = mn;
    }
    const float inv = 1.f / l;
    float o[3];
    mat3(Wo, a0 * inv, a1 * inv, a2 * inv, o);
    const float x0 = o[0] + bo[0], x1 = o[1] + bo[1], x2 = o[2] + bo[2];

    float t[3], d1[3], t2[3], d2[3], f[3];
    mat3(ext, x0, x1, x2, t);          // Wc
    mat3(U3b, t[0], t[1], t[2], d1);
    vnrelu1(t, d1);
    mat3(W3c, t[0], t[1], t[2], t2);
    mat3(U3c, t2[0], t2[1], t2[2], d2);
    vnrelu1(t2, d2);
    mat3(W3d, t2[0], t2[1], t2[2], f);
    const float xc0 = x0 + f[0], xc1 = x1 + f[1], xc2 = x2 + f[2];

    const float n0a = fabsf(xc0), n1a = fabsf(xc1), n2a = fabsf(xc2);
    const float mu  = (n0a + n1a + n2a) * (1.f / 3.f);
    const float var = (n0a * n0a + n1a * n1a + n2a * n2a) * (1.f / 3.f) - mu * mu;
    const float rst = 1.f / sqrtf(var + EPSV);
    float* op = xout + (size_t)row * 3;
    op[0] = xc0 / (n0a + EPSV) * ((n0a - mu) * rst * g2[0] + b2[0]);
    op[1] = xc1 / (n1a + EPSV) * ((n1a - mu) * rst * g2[1] + b2[1]);
    op[2] = xc2 / (n2a + EPSV) * ((n2a - mu) * rst * g2[2] + b2[2]);
  }
}

// ---------------------------------------------------------------------------
// Kernel B: h_cc = ref_feat(Mx128) @ We.T + be, fp32 WMMA.
// One 16-row strip per wave; A fragments register-resident (read A once);
// We in pad-132 LDS (conflict-free ds_load_b64), bias fused into the store.
// ---------------------------------------------------------------------------
__global__ void __launch_bounds__(256)
hcc_gemm_kernel(const float* __restrict__ A, const float* __restrict__ We,
                const float* __restrict__ be, float* __restrict__ Hout, int M) {
  extern __shared__ float wl[];                // 256*132 floats = 135,168 B
  const int tid  = threadIdx.x;
  const int lane = tid & 31;
  const int wave = tid >> 5;
  const int l15  = lane & 15;
  const int kp   = (lane >> 4) << 1;

  // Stage We with +4 float row pad (16B-aligned rows, bank-conflict free).
  for (int i4 = tid; i4 < 8192; i4 += 256) {
    const int f   = i4 << 2;
    const int row = f >> 7;
    const int kk  = f & 127;
    *(v2f*)(wl + row * WSTRIDE + kk)     = *(const v2f*)(We + f);
    *(v2f*)(wl + row * WSTRIDE + kk + 2) = *(const v2f*)(We + f + 2);
  }
  __syncthreads();

  const int nstrips = (M + 15) >> 4;
  const int gstride = gridDim.x * 8;
  for (int strip = blockIdx.x * 8 + wave; strip < nstrips; strip += gstride) {
    int arow = strip * 16 + l15;
    if (arow >= M) arow = M - 1;
    const float* ap = A + (size_t)arow * 128 + kp;
    __builtin_prefetch(A + ((size_t)arow + 16 * (size_t)gstride) * 128, 0, 1);
    v2f afrag[32];
#pragma unroll
    for (int kk = 0; kk < 32; ++kk) afrag[kk] = *(const v2f*)(ap + kk * 4);

    const int rbase = strip * 16 + ((lane >> 4) << 3);
    for (int ct = 0; ct < 16; ++ct) {
      const int colo = ct * 16 + l15;
      const float* bp = wl + colo * WSTRIDE + kp;
      v8f acc = {0.f, 0.f, 0.f, 0.f, 0.f, 0.f, 0.f, 0.f};
#pragma unroll
      for (int kk = 0; kk < 32; ++kk) {
        v2f b = *(const v2f*)(bp + kk * 4);
        acc = __builtin_amdgcn_wmma_f32_16x16x4_f32(false, afrag[kk], false, b,
                                                    (short)0, acc, false, false);
      }
      const float bias = be[colo];
#pragma unroll
      for (int r = 0; r < 8; ++r) {
        const int rowo = rbase + r;
        if (rowo < M) Hout[(size_t)rowo * 256 + colo] = acc[r] + bias;
      }
    }
  }
}

// ---------------------------------------------------------------------------
extern "C" void kernel_launch(void* const* d_in, const int* in_sizes, int n_in,
                              void* d_out, int out_size, void* d_ws, size_t ws_size,
                              hipStream_t stream) {
  const float* Z_V      = (const float*)d_in[0];
  const float* B_x      = (const float*)d_in[1];
  const float* ref_feat = (const float*)d_in[2];
  const float* Wff1 = (const float*)d_in[3];
  const float* W1   = (const float*)d_in[4];
  const float* U1   = (const float*)d_in[5];
  const float* W2   = (const float*)d_in[6];
  const float* U2   = (const float*)d_in[7];
  const float* W3   = (const float*)d_in[8];
  const float* g1   = (const float*)d_in[9];
  const float* b1   = (const float*)d_in[10];
  const float* Wq   = (const float*)d_in[11];
  const float* bq   = (const float*)d_in[12];
  const float* Wk   = (const float*)d_in[13];
  const float* bk   = (const float*)d_in[14];
  const float* Wv   = (const float*)d_in[15];
  const float* bv   = (const float*)d_in[16];
  const float* Wo   = (const float*)d_in[17];
  const float* bo   = (const float*)d_in[18];
  const float* W3a  = (const float*)d_in[19];
  const float* W3bm = (const float*)d_in[20];
  const float* U3b  = (const float*)d_in[21];
  const float* W3c  = (const float*)d_in[22];
  const float* U3c  = (const float*)d_in[23];
  const float* W3d  = (const float*)d_in[24];
  const float* g2   = (const float*)d_in[25];
  const float* b2   = (const float*)d_in[26];
  const float* We   = (const float*)d_in[27];
  const float* be   = (const float*)d_in[28];
  const int*   fg_idx = (const int*)d_in[29];

  const int N = in_sizes[0] / (FCH * 3);
  const int M = in_sizes[1] / 3;

  float* xout = (float*)d_out;                 // (M, 3)
  float* hout = xout + (size_t)M * 3;          // (M, 256)

  // bf16 weight staging in d_ws (element offsets; total 524,288 bf16 = 1MB).
  bf16* wsb   = (bf16*)d_ws;
  bf16* Wff1b = wsb;
  bf16* W1b   = wsb + 131072;
  bf16* U1b   = wsb + 262144;
  bf16* W2b   = wsb + 327680;
  bf16* U2b   = wsb + 393216;
  bf16* W3b_  = wsb + 458752;
  cvt_bf16_kernel<<<131072 / 256, 256, 0, stream>>>(Wff1, Wff1b, 131072);
  cvt_bf16_kernel<<<131072 / 256, 256, 0, stream>>>(W1,   W1b,   131072);
  cvt_bf16_kernel<<<65536 / 256,  256, 0, stream>>>(U1,   U1b,   65536);
  cvt_bf16_kernel<<<65536 / 256,  256, 0, stream>>>(W2,   W2b,   65536);
  cvt_bf16_kernel<<<65536 / 256,  256, 0, stream>>>(U2,   U2b,   65536);
  cvt_bf16_kernel<<<65536 / 256,  256, 0, stream>>>(W3,   W3b_,  65536);

  // Kernel A: 16 items/block, 150,016 B dynamic LDS -> 2 blocks/WGP.
  const size_t ldsA = (size_t)(37440 + 64) * sizeof(float);
  vn_decoder_kernel<<<N / NB, 256, ldsA, stream>>>(
      Z_V, B_x, Wff1b, W1b, U1b, W2b, U2b, W3b_, g1, b1,
      Wq, bq, Wk, bk, Wv, bv, Wo, bo,
      W3a, W3bm, U3b, W3c, U3c, W3d, g2, b2, fg_idx, xout);

  // Kernel B: one 16-row strip per wave.
  const int nstrips = (M + 15) / 16;
  const int blocksB = (nstrips + 7) / 8;
  const size_t ldsB = (size_t)(256 * WSTRIDE) * sizeof(float);
  hcc_gemm_kernel<<<blocksB, 256, ldsB, stream>>>(ref_feat, We, be, hout, M);
}
